// Attention_org_cross_85074712199395
// MI455X (gfx1250) — matrix operand order repaired and checked by reading, not compiled, not run on hardware
//
#include <hip/hip_runtime.h>

// Problem constants (match reference)
#define B_   4
#define N_   4096
#define C_   1024
#define H_   4
#define D_   256            // Ch == Kh
#define BH_  (B_ * H_)
#define BN_  (B_ * N_)
#define EPS_ 1e-5f

// ---- WMMA / TDM types (CDNA5 / gfx1250, wave32) ----
typedef __attribute__((ext_vector_type(16))) __bf16 v16bf;
typedef __attribute__((ext_vector_type(8)))  float  v8f;
typedef __attribute__((ext_vector_type(4)))  unsigned int v4u;
typedef __attribute__((ext_vector_type(8)))  int    v8i;
typedef __attribute__((ext_vector_type(4)))  int    v4i;

union BF16Frag {
  v16bf v;
  unsigned int   p[8];    // bf16 pairs (lo = even element)
  unsigned short u[16];
};

// Hardware packed convert: two f32 -> packed bf16 pair (RNE), one VALU op.
__device__ __forceinline__ unsigned int cvt_pk_bf16(float lo, float hi) {
  unsigned int r;
  asm volatile("v_cvt_pk_bf16_f32 %0, %1, %2" : "=v"(r) : "v"(lo), "v"(hi));
  return r;
}

__device__ __forceinline__ unsigned short f2bf1(float x) {
  return (unsigned short)cvt_pk_bf16(x, x);
}

__device__ __forceinline__ v8f wmma_bf16(const BF16Frag& a, const BF16Frag& b, v8f c) {
  return __builtin_amdgcn_wmma_f32_16x16x32_bf16(false, a.v, false, b.v,
                                                 (short)0, c, false, false);
}

// TDM: DMA one contiguous region (bytes, multiple of 4) from global into LDS
// offset 0 of the issuing wave's workgroup. D# per CDNA5 ISA:
// 2-D descriptor, data_size=4B, one row of `words` elements.
// clang-23 / amdgpu-toolchain form: 6 args (g0, g1, g2, g3, extra v8i, cpol).
__device__ __forceinline__ void tdm_load_contig_to_lds0(const void* gsrc,
                                                        unsigned words) {
  unsigned long long ga = (unsigned long long)gsrc;
  v4u g0;
  g0[0] = 1u;                                        // count=1, user descriptor
  g0[1] = 0u;                                        // lds_addr = 0
  g0[2] = (unsigned)(ga & 0xffffffffu);              // global_addr[31:0]
  g0[3] = (unsigned)((ga >> 32) & 0x01ffffffu)       // global_addr[56:32]
          | 0x80000000u;                             // type=2 ("image")
  v8i g1 = {};
  g1[0] = 0x00020000;                                // data_size=2 -> 4-byte elems
  g1[1] = (int)((words & 0xffffu) << 16);            // tensor_dim0[15:0] @bits63:48
  g1[2] = (int)((words >> 16) & 0xffffu)             // tensor_dim0[31:16] @bits79:64
          | (1 << 16);                               // tensor_dim1 = 1    @bits111:80
  g1[3] = (int)((words & 0xffffu) << 16);            // tile_dim0          @bits127:112
  g1[4] = 1;                                         // tile_dim1 = 1
  g1[5] = (int)words;                                // tensor_dim0_stride low32
  v4i gz  = {};
  v8i gz8 = {};
  __builtin_amdgcn_tensor_load_to_lds(g0, g1, gz, gz, gz8, 0);
}

// ---------------------------------------------------------------------------
// Fragment layouts (CDNA5 ISA, 16-bit A 16x32 / B 32x16, wave32):
//   A: lane L -> row m = L%16, half hi = L/16; pair v holds K = hi*16 + 2v (contiguous k order, same as B)
//   B: lane L -> col n = L%16, half c2 = L/16; pair v holds K = c2*16 + 2v
//   C/D f32:  lane L -> col = L%16; VGPR r -> row = r + 8*(L/16)
// Each wave computes a 32x32 tile (2x2 WMMA frags) for A/B register reuse.
// ---------------------------------------------------------------------------

// Projections. transposeOut==0: P[b,h,n,o] row-major (for V/Vd, feeds ctx GEMM A).
//              transposeOut==1: P[b,h,o,n] transposed  (for Q/Qd/K/Kd, feeds score
//              GEMM with contiguous-pair loads); the D-fragment's 8 consecutive
//              M-rows per lane become one 16-byte store in this layout.
__global__ void proj_kernel(const float* __restrict__ X,
                            const float* __restrict__ W,
                            unsigned short* __restrict__ P,
                            int wHeadStride, int transposeOut) {
  int lane = threadIdx.x & 31;
  int wave = threadIdx.x >> 5;
  int m0 = (blockIdx.x * 8 + wave) * 32;    // row tile over B*N (32 rows)
  int o0 = blockIdx.y * 32;                 // output-channel tile (32 cols)
  int h  = blockIdx.z;
  int lm = lane & 15, hi = lane >> 4;

  const float* Wh  = W + (size_t)h * wHeadStride;         // [D,D] (out,in)
  const float* xr0 = X + (size_t)(m0 + lm) * C_ + h * D_;
  const float* xr1 = xr0 + (size_t)16 * C_;
  const float* wr0 = Wh + (size_t)(o0 + lm) * D_;
  const float* wr1 = wr0 + (size_t)16 * D_;

  v8f a00 = {}, a01 = {}, a10 = {}, a11 = {};
  for (int k0 = 0; k0 < D_; k0 += 32) {
    BF16Frag fa0, fa1, fb0, fb1;
#pragma unroll
    for (int v = 0; v < 8; ++v) {
      int kk = k0 + (hi << 4) + 2 * v;
      float2 t0 = *(const float2*)(xr0 + kk);
      float2 t1 = *(const float2*)(xr1 + kk);
      fa0.p[v] = cvt_pk_bf16(t0.x, t0.y);
      fa1.p[v] = cvt_pk_bf16(t1.x, t1.y);
    }
#pragma unroll
    for (int v = 0; v < 8; ++v) {
      int kk = k0 + (hi << 4) + 2 * v;
      float2 t0 = *(const float2*)(wr0 + kk);
      float2 t1 = *(const float2*)(wr1 + kk);
      fb0.p[v] = cvt_pk_bf16(t0.x, t0.y);
      fb1.p[v] = cvt_pk_bf16(t1.x, t1.y);
    }
    a00 = wmma_bf16(fa0, fb0, a00);
    a01 = wmma_bf16(fa0, fb1, a01);
    a10 = wmma_bf16(fa1, fb0, a10);
    a11 = wmma_bf16(fa1, fb1, a11);
  }

  if (transposeOut) {
#pragma unroll
    for (int i = 0; i < 2; ++i) {
      v8f accL = i ? a10 : a00;
      v8f accR = i ? a11 : a01;
      int mr = m0 + i * 16 + (hi << 3);       // first of 8 consecutive rows
      int b = mr >> 12;
      int n = mr & (N_ - 1);
      size_t baseL = (((size_t)b * H_ + h) * D_ + (o0 + lm)) * N_ + n;
      size_t baseR = (((size_t)b * H_ + h) * D_ + (o0 + 16 + lm)) * N_ + n;
      uint4 tL, tR;
      tL.x = cvt_pk_bf16(accL[0], accL[1]); tL.y = cvt_pk_bf16(accL[2], accL[3]);
      tL.z = cvt_pk_bf16(accL[4], accL[5]); tL.w = cvt_pk_bf16(accL[6], accL[7]);
      tR.x = cvt_pk_bf16(accR[0], accR[1]); tR.y = cvt_pk_bf16(accR[2], accR[3]);
      tR.z = cvt_pk_bf16(accR[4], accR[5]); tR.w = cvt_pk_bf16(accR[6], accR[7]);
      *(uint4*)(P + baseL) = tL;              // global_store_b128
      *(uint4*)(P + baseR) = tR;
    }
  } else {
#pragma unroll
    for (int i = 0; i < 2; ++i) {
      v8f accL = i ? a10 : a00;
      v8f accR = i ? a11 : a01;
#pragma unroll
      for (int r = 0; r < 8; ++r) {
        int mr = m0 + i * 16 + r + (hi << 3);
        int b = mr >> 12;
        int n = mr & (N_ - 1);
        size_t base = (((size_t)b * H_ + h) * N_ + n) * D_;
        P[base + o0 + lm]      = f2bf1(accL[r]);
        P[base + o0 + 16 + lm] = f2bf1(accR[r]);
      }
    }
  }
}

// S[bh,c,k] = scale * sum_n QT[bh,c,n] * KT[bh,k,n]   (both operands transposed ->
// all fragment loads are contiguous bf16 pairs)
__global__ void score_kernel(const unsigned short* __restrict__ QT,  // [BH,D,N]
                             const unsigned short* __restrict__ KT,  // [BH,D,N]
                             float* __restrict__ S, float scale) {
  int lane = threadIdx.x & 31;
  int wave = threadIdx.x >> 5;
  int bh = blockIdx.z;
  int m0 = (blockIdx.x * 4 + wave) * 32;   // c (query-channel) tile
  int c0 = blockIdx.y * 32;                // k (key-channel) tile
  int lm = lane & 15, hi = lane >> 4;

  const unsigned short* ar0 = QT + (size_t)bh * D_ * N_ + (size_t)(m0 + lm) * N_;
  const unsigned short* ar1 = ar0 + (size_t)16 * N_;
  const unsigned short* br0 = KT + (size_t)bh * D_ * N_ + (size_t)(c0 + lm) * N_;
  const unsigned short* br1 = br0 + (size_t)16 * N_;

  v8f a00 = {}, a01 = {}, a10 = {}, a11 = {};
  for (int n0 = 0; n0 < N_; n0 += 32) {
    BF16Frag fa0, fa1, fb0, fb1;
#pragma unroll
    for (int v = 0; v < 8; ++v) {          // A[c, n] = QT[c, n]
      int kk = n0 + (hi << 4) + 2 * v;
      fa0.p[v] = *(const unsigned int*)(ar0 + kk);
      fa1.p[v] = *(const unsigned int*)(ar1 + kk);
    }
#pragma unroll
    for (int v = 0; v < 8; ++v) {          // B[n, k] = KT[k, n]
      int kk = n0 + (hi << 4) + 2 * v;
      fb0.p[v] = *(const unsigned int*)(br0 + kk);
      fb1.p[v] = *(const unsigned int*)(br1 + kk);
    }
    a00 = wmma_bf16(fa0, fb0, a00);
    a01 = wmma_bf16(fa0, fb1, a01);
    a10 = wmma_bf16(fa1, fb0, a10);
    a11 = wmma_bf16(fa1, fb1, a11);
  }
  float* Sp = S + (size_t)bh * D_ * D_;
  int kc0 = c0 + lm, kc1 = kc0 + 16;
#pragma unroll
  for (int i = 0; i < 2; ++i) {
    v8f accL = i ? a10 : a00;
    v8f accR = i ? a11 : a01;
#pragma unroll
    for (int r = 0; r < 8; ++r) {
      int c = m0 + i * 16 + r + (hi << 3);
      Sp[(size_t)c * D_ + kc0] = accL[r] * scale;
      Sp[(size_t)c * D_ + kc1] = accR[r] * scale;
    }
  }
}

// InstanceNorm2d(affine=False) over each 256x256 (c,k) plane, then row softmax.
__global__ void inorm_softmax_kernel(const float* __restrict__ S,
                                     unsigned short* __restrict__ P) {
  int bh = blockIdx.x;
  const float* s = S + (size_t)bh * D_ * D_;
  __shared__ float red[256];

  float sum = 0.f, sq = 0.f;
  for (int i = threadIdx.x; i < D_ * D_; i += 256) {
    float x = s[i]; sum += x; sq += x * x;
  }
  red[threadIdx.x] = sum; __syncthreads();
  for (int o = 128; o > 0; o >>= 1) {
    if ((int)threadIdx.x < o) red[threadIdx.x] += red[threadIdx.x + o];
    __syncthreads();
  }
  float mean = red[0] * (1.f / (D_ * D_));
  __syncthreads();
  red[threadIdx.x] = sq; __syncthreads();
  for (int o = 128; o > 0; o >>= 1) {
    if ((int)threadIdx.x < o) red[threadIdx.x] += red[threadIdx.x + o];
    __syncthreads();
  }
  float var  = red[0] * (1.f / (D_ * D_)) - mean * mean;
  float rstd = rsqrtf(var + EPS_);

  const float* row = s + (size_t)threadIdx.x * D_;
  float mx = -3.402823e38f;
  for (int k = 0; k < D_; ++k)
    mx = fmaxf(mx, (row[k] - mean) * rstd);
  float se = 0.f;
  for (int k = 0; k < D_; ++k)
    se += __expf((row[k] - mean) * rstd - mx);
  float inv = 1.f / se;
  unsigned int* prow =
      (unsigned int*)(P + (size_t)bh * D_ * D_ + (size_t)threadIdx.x * D_);
  for (int k = 0; k < D_; k += 2) {
    float e0 = __expf((row[k]     - mean) * rstd - mx) * inv;
    float e1 = __expf((row[k + 1] - mean) * rstd - mx) * inv;
    prow[k >> 1] = cvt_pk_bf16(e0, e1);   // packed bf16 store
  }
}

// Ctx[b, n, c*H + h] = sum_k Pm[bh, c, k] * V[bh, n, k]
// The 32-row x 256-k probs B-tile (16 KB, contiguous) is DMA'd once per block
// into LDS by the Tensor Data Mover; all 4 waves then read fragments from LDS.
__global__ void ctx_kernel(const unsigned short* __restrict__ V,   // [BH,N,D]
                           const unsigned short* __restrict__ Pm,  // [BH,D,D]
                           unsigned short* __restrict__ CtxO) {    // [B,N,C]
  __shared__ unsigned short plds[32 * D_];   // 16 KB probs tile, at LDS offset 0

  int lane = threadIdx.x & 31;
  int wave = threadIdx.x >> 5;
  int bh = blockIdx.z;
  int n0 = (blockIdx.x * 4 + wave) * 32;   // token tile (M)
  int c0 = blockIdx.y * 32;                // channel tile (N)
  int lm = lane & 15, hi = lane >> 4;

  if (wave == 0) {                          // one TDM DMA for the whole B-tile
    const void* btile = Pm + (size_t)bh * D_ * D_ + (size_t)c0 * D_;
    tdm_load_contig_to_lds0(btile, (32 * D_ * 2) / 4);   // 4096 dwords
    __builtin_amdgcn_s_wait_tensorcnt(0);
  }
  __syncthreads();

  const unsigned short* ar0 = V + (size_t)bh * N_ * D_ + (size_t)(n0 + lm) * D_;
  const unsigned short* ar1 = ar0 + (size_t)16 * D_;
  const unsigned int* bl0 = (const unsigned int*)(plds + (size_t)lm * D_);
  const unsigned int* bl1 = (const unsigned int*)(plds + (size_t)(16 + lm) * D_);

  v8f a00 = {}, a01 = {}, a10 = {}, a11 = {};
  for (int k0 = 0; k0 < D_; k0 += 32) {
    BF16Frag fa0, fa1, fb0, fb1;
#pragma unroll
    for (int v = 0; v < 8; ++v) {          // A[n, k] = V[n, k]
      int kk = k0 + (hi << 4) + 2 * v;
      fa0.p[v] = *(const unsigned int*)(ar0 + kk);
      fa1.p[v] = *(const unsigned int*)(ar1 + kk);
    }
#pragma unroll
    for (int v = 0; v < 8; ++v) {          // B[k, c] = probs from LDS (ds_load)
      int kp = (k0 >> 1) + (hi << 3) + v;  // pair index within row
      fb0.p[v] = bl0[kp];
      fb1.p[v] = bl1[kp];
    }
    a00 = wmma_bf16(fa0, fb0, a00);
    a01 = wmma_bf16(fa0, fb1, a01);
    a10 = wmma_bf16(fa1, fb0, a10);
    a11 = wmma_bf16(fa1, fb1, a11);
  }
  int b = bh >> 2, h = bh & 3;
#pragma unroll
  for (int i = 0; i < 2; ++i) {
    v8f accL = i ? a10 : a00;
    v8f accR = i ? a11 : a01;
#pragma unroll
    for (int r = 0; r < 8; ++r) {
      int n = n0 + i * 16 + r + (hi << 3);
      size_t base = ((size_t)b * N_ + n) * C_;
      CtxO[base + (size_t)(c0 + lm) * H_ + h]      = f2bf1(accL[r]);   // c-major, h-minor
      CtxO[base + (size_t)(c0 + 16 + lm) * H_ + h] = f2bf1(accR[r]);
    }
  }
}

// O[m, o] = sum_c Ctx[m, c] * W[o, c]   (dominant GEMM; weight tile staged in LDS,
// shared by all 8 waves of the block -> 8x less global weight traffic)
__global__ void out_kernel(const unsigned short* __restrict__ Ctx,  // [BN, C] bf16
                           const float* __restrict__ W,             // [C, C] f32
                           float* __restrict__ O) {                 // [BN, C] f32
  int lane = threadIdx.x & 31;
  int wave = threadIdx.x >> 5;
  int m0 = (blockIdx.x * 8 + wave) * 32;
  int o0 = blockIdx.y * 32;
  int lm = lane & 15, hi = lane >> 4;

  // LDS weight tile: 32 o-columns x 32 k, bf16 pairs -> [o][kpair], 2 KB
  __shared__ unsigned int wlds[32 * 16];

  const unsigned short* ar0 = Ctx + (size_t)(m0 + lm) * C_;
  const unsigned short* ar1 = ar0 + (size_t)16 * C_;
  int to = threadIdx.x >> 3;               // 0..31  (staging: o-column)
  int tk = threadIdx.x & 7;                // 0..7   (staging: float4 within row)
  const float* wstg = W + (size_t)(o0 + to) * C_ + tk * 4;

  v8f a00 = {}, a01 = {}, a10 = {}, a11 = {};
  for (int k0 = 0; k0 < C_; k0 += 32) {
    __builtin_prefetch(ar0 + k0 + 128, 0, 1);    // global_prefetch_b8
    __syncthreads();                             // previous tile fully consumed
    float4 wv = *(const float4*)(wstg + k0);     // coalesced f32 weight load
    wlds[to * 16 + tk * 2]     = cvt_pk_bf16(wv.x, wv.y);
    wlds[to * 16 + tk * 2 + 1] = cvt_pk_bf16(wv.z, wv.w);
    __syncthreads();

    BF16Frag fa0, fa1, fb0, fb1;
#pragma unroll
    for (int v = 0; v < 8; ++v) {
      int kk = k0 + (hi << 4) + 2 * v;
      fa0.p[v] = *(const unsigned int*)(ar0 + kk);
      fa1.p[v] = *(const unsigned int*)(ar1 + kk);
    }
#pragma unroll
    for (int v = 0; v < 8; ++v) {                // B frags from LDS (ds_load)
      fb0.p[v] = wlds[lm * 16 + (hi << 3) + v];
      fb1.p[v] = wlds[(16 + lm) * 16 + (hi << 3) + v];
    }
    a00 = wmma_bf16(fa0, fb0, a00);
    a01 = wmma_bf16(fa0, fb1, a01);
    a10 = wmma_bf16(fa1, fb0, a10);
    a11 = wmma_bf16(fa1, fb1, a11);
  }
#pragma unroll
  for (int i = 0; i < 2; ++i) {
    v8f accL = i ? a10 : a00;
    v8f accR = i ? a11 : a01;
#pragma unroll
    for (int r = 0; r < 8; ++r) {
      int m = m0 + i * 16 + r + (hi << 3);
      O[(size_t)m * C_ + o0 + lm]      = accL[r];
      O[(size_t)m * C_ + o0 + 16 + lm] = accR[r];
    }
  }
}

extern "C" void kernel_launch(void* const* d_in, const int* in_sizes, int n_in,
                              void* d_out, int out_size, void* d_ws, size_t ws_size,
                              hipStream_t stream) {
  (void)in_sizes; (void)n_in; (void)out_size; (void)ws_size;

  const float* emb1     = (const float*)d_in[0];
  const float* emb_all  = (const float*)d_in[1];
  const float* embd1    = (const float*)d_in[2];
  const float* emb_alld = (const float*)d_in[3];
  const float* Wq    = (const float*)d_in[4];
  const float* Wqd   = (const float*)d_in[5];
  const float* Wk0   = (const float*)d_in[6];
  const float* Wv0   = (const float*)d_in[7];
  const float* Wkd0  = (const float*)d_in[8];
  const float* Wvd0  = (const float*)d_in[9];
  const float* Wout  = (const float*)d_in[10];
  const float* Woutd = (const float*)d_in[11];
  float* out = (float*)d_out;

  // Workspace layout (bf16 intermediates; ~280 MB total)
  const size_t PROJ_B = (size_t)BH_ * N_ * D_ * sizeof(unsigned short);  // 32 MiB
  const size_t S_B    = (size_t)BH_ * D_ * D_ * sizeof(float);           // 4 MiB
  const size_t P_B    = (size_t)BH_ * D_ * D_ * sizeof(unsigned short);  // 2 MiB
  const size_t CTX_B  = (size_t)BN_ * C_ * sizeof(unsigned short);       // 32 MiB

  char* ws = (char*)d_ws; size_t off = 0;
  auto take = [&](size_t bytes) { void* p = ws + off; off += (bytes + 255) & ~(size_t)255; return p; };
  unsigned short* QTp  = (unsigned short*)take(PROJ_B);   // [BH,D,N] transposed
  unsigned short* QdTp = (unsigned short*)take(PROJ_B);   // [BH,D,N]
  unsigned short* KTp  = (unsigned short*)take(PROJ_B);   // [BH,D,N]
  unsigned short* Vp   = (unsigned short*)take(PROJ_B);   // [BH,N,D] row-major
  unsigned short* KdTp = (unsigned short*)take(PROJ_B);   // [BH,D,N]
  unsigned short* Vdp  = (unsigned short*)take(PROJ_B);   // [BH,N,D]
  float*          S1   = (float*)take(S_B);
  float*          Sd1  = (float*)take(S_B);
  unsigned short* P1   = (unsigned short*)take(P_B);
  unsigned short* Pd1  = (unsigned short*)take(P_B);
  unsigned short* Ctx1 = (unsigned short*)take(CTX_B);
  unsigned short* Ctxd = (unsigned short*)take(CTX_B);

  // 1) Projections. Q/Qd/K/Kd stored transposed [bh,o,n] (b128 stores, feeds
  //    score GEMM with contiguous loads); V/Vd row-major (feeds ctx GEMM).
  dim3 bp(256), gp(BN_ / 32 / 8, D_ / 32, H_);   // (64,8,4)
  proj_kernel<<<gp, bp, 0, stream>>>(emb1,     Wq,   QTp,  D_ * D_, 1);
  proj_kernel<<<gp, bp, 0, stream>>>(embd1,    Wqd,  QdTp, D_ * D_, 1);
  proj_kernel<<<gp, bp, 0, stream>>>(emb_all,  Wk0,  KTp,  0,       1);
  proj_kernel<<<gp, bp, 0, stream>>>(emb_all,  Wv0,  Vp,   0,       0);
  proj_kernel<<<gp, bp, 0, stream>>>(emb_alld, Wkd0, KdTp, 0,       1);
  proj_kernel<<<gp, bp, 0, stream>>>(emb_alld, Wvd0, Vdp,  0,       0);

  // 2) Channel-attention scores (contract over 4096 tokens).
  const float scale = 0.03125f;                  // 1/sqrt(1024)
  dim3 bs(128), gs(D_ / 32 / 4, D_ / 32, BH_);   // (2,8,16)
  score_kernel<<<gs, bs, 0, stream>>>(QTp,  KdTp, S1,  scale);
  score_kernel<<<gs, bs, 0, stream>>>(QdTp, KTp,  Sd1, scale);

  // 3) InstanceNorm + softmax per (b,h) plane.
  inorm_softmax_kernel<<<dim3(BH_), dim3(256), 0, stream>>>(S1,  P1);
  inorm_softmax_kernel<<<dim3(BH_), dim3(256), 0, stream>>>(Sd1, Pd1);

  // 4) Context (probs x V^T); probs tile TDM-DMA'd to LDS once per block.
  dim3 bc(128), gc(N_ / 32 / 4, D_ / 32, BH_);   // (32,8,16)
  ctx_kernel<<<gc, bc, 0, stream>>>(Vdp, P1,  Ctx1);
  ctx_kernel<<<gc, bc, 0, stream>>>(Vp,  Pd1, Ctxd);

  // 5) Output projections (dominant GEMMs, 1024-deep K, LDS-staged weights).
  dim3 bo(256), go(BN_ / 32 / 8, C_ / 32);       // (64,32)
  out_kernel<<<go, bo, 0, stream>>>(Ctx1, Wout,  out);
  out_kernel<<<go, bo, 0, stream>>>(Ctxd, Woutd, out + (size_t)BN_ * C_);
}
